// InstanceMemory_26826365731330
// MI455X (gfx1250) — compile-verified, matrix operand order checked by reading, not run on hardware
//
#include <hip/hip_runtime.h>
#include <hip/hip_bf16.h>

typedef _Float16 v8h  __attribute__((ext_vector_type(8)));
typedef _Float16 v16h __attribute__((ext_vector_type(16)));
typedef float    v4f  __attribute__((ext_vector_type(4)));
typedef float    v8f  __attribute__((ext_vector_type(8)));

#define D_K     2048
#define N_MEM   49152
#define B_ROWS  256
#define INV_T   20.0f

// ---------------------------------------------------------------------------
// Kernel 1: row L2-normalize inputs / inputs_ema -> f16 copies
// grid 512 (rows 0..255 = inputs, 256..511 = inputs_ema), block 256
// ---------------------------------------------------------------------------
__global__ __launch_bounds__(256) void norm_k(const float* __restrict__ x,
                                              const float* __restrict__ xe,
                                              _Float16* __restrict__ xh,
                                              _Float16* __restrict__ xeh) {
  __shared__ float red[256];
  __shared__ float sc;
  const int row = blockIdx.x;
  const float*    src = (row < B_ROWS) ? x  : xe;
  _Float16*       dst = (row < B_ROWS) ? xh : xeh;
  const int r   = row & (B_ROWS - 1);
  const int tid = threadIdx.x;

  const float* p = src + (size_t)r * D_K + tid * 8;
  float v[8];
  float s = 0.f;
#pragma unroll
  for (int j = 0; j < 8; ++j) { v[j] = p[j]; s += v[j] * v[j]; }
  red[tid] = s;
  __syncthreads();
  for (int st = 128; st > 0; st >>= 1) {
    if (tid < st) red[tid] += red[tid + st];
    __syncthreads();
  }
  if (tid == 0) sc = 1.0f / fmaxf(sqrtf(red[0]), 1e-12f);
  __syncthreads();
  const float scale = sc;
  _Float16* q = dst + (size_t)r * D_K + tid * 8;
#pragma unroll
  for (int j = 0; j < 8; ++j) q[j] = (_Float16)(v[j] * scale);
}

// ---------------------------------------------------------------------------
// Kernel 2: WMMA GEMM  out[M=256][Nn] = A(f16)[256][2048] * B[Nn][2048]^T
// B source is f32 (features, converted through LDS) or f16 (xeh).
// Block: 256 threads (8 waves). Block tile: full M=256 x NT=64 columns.
// Wave w: rows [32w, 32w+32), all 64 cols -> 2x4 tiles of 16x16 (64 acc VGPRs).
// ---------------------------------------------------------------------------
template <bool B_IS_F32>
__global__ __launch_bounds__(256) void gemm_k(const _Float16* __restrict__ A,
                                              const void* __restrict__ Bsrc,
                                              float* __restrict__ out, int Nn) {
  constexpr int KT = 32;      // K step per iteration
  constexpr int NT = 64;      // columns per block
  constexpr int LDB = 40;     // padded LDS row stride in halves (80B, 16B-aligned)
  __shared__ _Float16 Bs[NT * LDB];

  const int tid  = threadIdx.x;
  const int wave = tid >> 5;
  const int lane = tid & 31;
  const int lm   = lane & 15;   // tile row/col within 16
  const int lh   = lane >> 4;   // lane half selects K sub-chunk
  const int nblk = blockIdx.x * NT;

  const int srow  = tid >> 2;        // staging: 4 threads per B row
  const int skoff = (tid & 3) * 8;   // 8 halves per thread

  v8f acc[2][4] = {};

  for (int k0 = 0; k0 < D_K; k0 += KT) {
    // ---- stage B tile (64 rows x 32 K) into LDS as f16 ----
    if constexpr (B_IS_F32) {
      const float* Bf = (const float*)Bsrc;
      const v4f* p = (const v4f*)(Bf + (size_t)(nblk + srow) * D_K + k0 + skoff);
      v4f a = p[0], b = p[1];
      v8h h;
      h[0] = (_Float16)a[0]; h[1] = (_Float16)a[1];
      h[2] = (_Float16)a[2]; h[3] = (_Float16)a[3];
      h[4] = (_Float16)b[0]; h[5] = (_Float16)b[1];
      h[6] = (_Float16)b[2]; h[7] = (_Float16)b[3];
      *(v8h*)(&Bs[srow * LDB + skoff]) = h;
    } else {
      const _Float16* Bh = (const _Float16*)Bsrc;
      *(v8h*)(&Bs[srow * LDB + skoff]) =
          *(const v8h*)(Bh + (size_t)(nblk + srow) * D_K + k0 + skoff);
    }
    __syncthreads();

    // ---- load fragments ----
    // A 16x32 f16 layout: lane<16 holds row=lm, K = k0..k0+7 (v0-3) and
    // k0+16..k0+23 (v4-7); lane>=16 holds K offset +8. Two 16B chunks.
    union { v16h v; v8h h[2]; } af[2], bf[4];
#pragma unroll
    for (int mt = 0; mt < 2; ++mt) {
      const _Float16* ap =
          A + (size_t)(wave * 32 + mt * 16 + lm) * D_K + k0 + lh * 8;
      af[mt].h[0] = *(const v8h*)ap;
      af[mt].h[1] = *(const v8h*)(ap + 16);
    }
#pragma unroll
    for (int nt = 0; nt < 4; ++nt) {
      const _Float16* bp = &Bs[(nt * 16 + lm) * LDB + lh * 8];
      bf[nt].h[0] = *(const v8h*)bp;
      bf[nt].h[1] = *(const v8h*)(bp + 16);
    }

#pragma unroll
    for (int mt = 0; mt < 2; ++mt)
#pragma unroll
      for (int nt = 0; nt < 4; ++nt)
        acc[mt][nt] = __builtin_amdgcn_wmma_f32_16x16x32_f16(
            false, af[mt].v, false, bf[nt].v, (short)0, acc[mt][nt],
            false, false);
    __syncthreads();
  }

  // ---- epilogue: C layout -> VGPR i: lanes0-15 M=i, lanes16-31 M=8+i; N=lm
#pragma unroll
  for (int mt = 0; mt < 2; ++mt)
#pragma unroll
    for (int nt = 0; nt < 4; ++nt) {
      const int col = nblk + nt * 16 + lm;
#pragma unroll
      for (int i = 0; i < 8; ++i) {
        const int row = wave * 32 + mt * 16 + lh * 8 + i;
        out[(size_t)row * Nn + col] = acc[mt][nt][i];
      }
    }
}

// ---------------------------------------------------------------------------
// Kernel 3: per row, exact sum of top-256 exp(sim/T) excluding the 16-column
// identity block. 3-level histogram radix select on monotone f32 keys
// (integer atomics only -> deterministic), then one summation pass.
// grid 256 (one per batch row), block 256
// ---------------------------------------------------------------------------
__device__ __forceinline__ unsigned fkey(float v) {
  unsigned u = __float_as_uint(v);
  return (u & 0x80000000u) ? ~u : (u | 0x80000000u);
}

__global__ __launch_bounds__(256) void topk_sum_k(const float* __restrict__ sims,
                                                  const int* __restrict__ targets,
                                                  float* __restrict__ sim_neg) {
  __shared__ unsigned h1[2048];
  __shared__ unsigned h2[2048];
  __shared__ unsigned c3[1024];
  __shared__ float    red[256];
  __shared__ unsigned s_t1, s_m1, s_t2, s_m2, s_kt, s_r;

  const int b   = blockIdx.x;
  const int tid = threadIdx.x;
  const int m0  = targets[b] * 16;           // masked columns [m0, m0+16)
  const float* row = sims + (size_t)b * N_MEM;

  // pass 1: top-11-bit histogram
  for (int i = tid; i < 2048; i += 256) h1[i] = 0;
  __syncthreads();
  for (int n = tid; n < N_MEM; n += 256) {
    if (n >= m0 && n < m0 + 16) continue;
    atomicAdd(&h1[fkey(row[n]) >> 21], 1u);
  }
  __syncthreads();
  if (tid == 0) {
    unsigned c = 0, i = 2047;
    for (;; --i) { if (c + h1[i] >= 256u || i == 0u) break; c += h1[i]; }
    s_t1 = i; s_m1 = c;
  }
  __syncthreads();
  const unsigned t1 = s_t1;

  // pass 2: middle-11-bit histogram within bin t1
  for (int i = tid; i < 2048; i += 256) h2[i] = 0;
  __syncthreads();
  for (int n = tid; n < N_MEM; n += 256) {
    if (n >= m0 && n < m0 + 16) continue;
    unsigned k = fkey(row[n]);
    if ((k >> 21) == t1) atomicAdd(&h2[(k >> 10) & 2047u], 1u);
  }
  __syncthreads();
  if (tid == 0) {
    unsigned c = s_m1, i = 2047;
    for (;; --i) { if (c + h2[i] >= 256u || i == 0u) break; c += h2[i]; }
    s_t2 = i; s_m2 = c;
  }
  __syncthreads();
  const unsigned t2 = s_t2;

  // pass 3: low-10-bit histogram (exact keys) within (t1,t2)
  for (int i = tid; i < 1024; i += 256) c3[i] = 0;
  __syncthreads();
  for (int n = tid; n < N_MEM; n += 256) {
    if (n >= m0 && n < m0 + 16) continue;
    unsigned k = fkey(row[n]);
    if ((k >> 21) == t1 && ((k >> 10) & 2047u) == t2)
      atomicAdd(&c3[k & 1023u], 1u);
  }
  __syncthreads();
  if (tid == 0) {
    unsigned c = s_m2, i = 1023;
    for (;; --i) { if (c + c3[i] >= 256u || i == 0u) break; c += c3[i]; }
    s_kt = (t1 << 21) | (t2 << 10) | i;   // exact 256th-largest key
    s_r  = 256u - c;                      // multiplicity taken at threshold
  }
  __syncthreads();
  const unsigned kt = s_kt;

  // pass 4: deterministic sum of exp for keys strictly above threshold
  float s = 0.f;
  for (int n = tid; n < N_MEM; n += 256) {
    if (n >= m0 && n < m0 + 16) continue;
    float v = row[n];
    if (fkey(v) > kt) s += expf(v * INV_T);
  }
  red[tid] = s;
  __syncthreads();
  for (int st = 128; st > 0; st >>= 1) {
    if (tid < st) red[tid] += red[tid + st];
    __syncthreads();
  }
  if (tid == 0) {
    unsigned u = (kt & 0x80000000u) ? (kt ^ 0x80000000u) : ~kt;
    float vt = __uint_as_float(u);
    sim_neg[b] = red[0] + (float)s_r * expf(vt * INV_T);
  }
}

// ---------------------------------------------------------------------------
// Kernel 4: sim_pos[b] = exp(min over same-target c of batch[b][c] / T)
// ---------------------------------------------------------------------------
__global__ __launch_bounds__(256) void posmin_k(const float* __restrict__ batch,
                                                const int* __restrict__ targets,
                                                float* __restrict__ sim_pos) {
  __shared__ int tg[256];
  const int tid = threadIdx.x;
  tg[tid] = targets[tid];
  __syncthreads();
  const int t = tg[tid];
  float m = 3.0e38f;
  for (int c = 0; c < 256; ++c)
    if (tg[c] == t) m = fminf(m, batch[tid * 256 + c]);
  sim_pos[tid] = expf(m * INV_T);
}

// ---------------------------------------------------------------------------
// Kernel 5: loss = -mean(log(pos/(pos+neg+1e-6) + 1e-6))
// ---------------------------------------------------------------------------
__global__ __launch_bounds__(256) void loss_k(const float* __restrict__ sim_pos,
                                              const float* __restrict__ sim_neg,
                                              float* __restrict__ out) {
  __shared__ float red[256];
  const int tid = threadIdx.x;
  const float p = sim_pos[tid];
  const float d = p + sim_neg[tid] + 1e-6f;
  red[tid] = -logf(p / d + 1e-6f);
  __syncthreads();
  for (int st = 128; st > 0; st >>= 1) {
    if (tid < st) red[tid] += red[tid + st];
    __syncthreads();
  }
  if (tid == 0) out[0] = red[0] * (1.0f / 256.0f);
}

// ---------------------------------------------------------------------------
extern "C" void kernel_launch(void* const* d_in, const int* in_sizes, int n_in,
                              void* d_out, int out_size, void* d_ws, size_t ws_size,
                              hipStream_t stream) {
  const float* inputs     = (const float*)d_in[0];
  const float* inputs_ema = (const float*)d_in[1];
  const float* features   = (const float*)d_in[2];
  const int*   targets    = (const int*)d_in[3];
  float*       out        = (float*)d_out;

  char* ws = (char*)d_ws;
  _Float16* xh      = (_Float16*)(ws);                                  // 1 MB
  _Float16* xeh     = (_Float16*)(ws + (1u << 20));                     // 1 MB
  float*    sims    = (float*)(ws + (2u << 20));                        // 50.33 MB
  float*    batch   = (float*)(ws + (2u << 20) + 50331648u);            // 256 KB
  float*    sim_pos = (float*)(ws + (2u << 20) + 50331648u + 262144u);  // 1 KB
  float*    sim_neg = sim_pos + B_ROWS;                                 // 1 KB

  norm_k<<<512, 256, 0, stream>>>(inputs, inputs_ema, xh, xeh);
  gemm_k<true ><<<N_MEM / 64, 256, 0, stream>>>(xh, (const void*)features, sims, N_MEM);
  gemm_k<false><<<B_ROWS / 64, 256, 0, stream>>>(xh, (const void*)xeh, batch, B_ROWS);
  topk_sum_k<<<B_ROWS, 256, 0, stream>>>(sims, targets, sim_neg);
  posmin_k<<<1, 256, 0, stream>>>(batch, targets, sim_pos);
  loss_k<<<1, 256, 0, stream>>>(sim_pos, sim_neg, out);
}
